// GGNNProper_16492674417501
// MI455X (gfx1250) — matrix operand order repaired and checked by reading, not compiled, not run on hardware
//
#include <hip/hip_runtime.h>
#include <hip/hip_bf16.h>
#include <math.h>

// ---------------------------------------------------------------------------
// GGNN on gfx1250: bf16 WMMA GEMMs (v_wmma_f32_16x16x32_bf16) + f32 atomics
// ---------------------------------------------------------------------------

#define NN     50000      // nodes
#define HH     202        // hidden
#define HP     224        // padded hidden (multiple of 32 for K of bf16 WMMA)
#define EE     100000     // edges per forward type
#define ETYPES 6          // 3 forward + 3 backward
#define CHUNK  208        // per-edge-type padded column chunk (multiple of 16)
#define PCOLS  (ETYPES * CHUNK)   // 1248, prop columns (padded)
#define GCOLS  608        // 3*202=606 -> 608 padded gi/gh columns
#define POSMAX 1024

typedef __bf16 bf16_t;
typedef __attribute__((ext_vector_type(16))) __bf16 v16bf;
typedef __attribute__((ext_vector_type(8)))  __bf16 v8bf;
typedef __attribute__((ext_vector_type(8)))  float  v8f;

union BF16x16 { v16bf v; v8bf h[2]; };

// A fragment, 16x32 bf16, row-major source with stride lda (in elements).
// Layout: lane<16 -> row=lane, K = k0+{0..15}
//         lane>=16 -> row=lane-16, K = k0+{16..31}
__device__ __forceinline__ v16bf load_a_frag(const bf16_t* __restrict__ A,
                                             int row0, int lda, int k0, int lane) {
  int r  = lane & 15;
  int kb = k0 + ((lane >> 4) << 4);               // +0 or +16
  const bf16_t* p = A + (size_t)(row0 + r) * lda + kb;
  BF16x16 f;
  f.h[0] = *(const v8bf*)(p);
  f.h[1] = *(const v8bf*)(p + 8);
  return f.v;
}

// B fragment, 32x16 bf16, from TRANSPOSED storage Bt[col][k] stride ldb.
// Layout used here (same k order as A): lanes 0-15 hold K=k0+0..15 of column N=lane; lanes 16-31 hold
// K=k0+16..31 of column N=lane-16.  Contiguous 16 bf16 per lane in Bt.
__device__ __forceinline__ v16bf load_bt_frag(const bf16_t* __restrict__ Bt,
                                              int col0, int ldb, int k0, int lane) {
  int c  = col0 + (lane & 15);
  int kb = k0 + ((lane >> 4) << 4);               // +0 or +16
  const bf16_t* p = Bt + (size_t)c * ldb + kb;
  BF16x16 f;
  f.h[0] = *(const v8bf*)(p);
  f.h[1] = *(const v8bf*)(p + 8);
  return f.v;
}

__device__ __forceinline__ v8f wmma_bf16(v16bf a, v16bf b, v8f c) {
  return __builtin_amdgcn_wmma_f32_16x16x32_bf16(false, a, false, b,
                                                 (short)0, c, false, false);
}

__device__ __forceinline__ float sigmoidf_(float x) {
  return 1.0f / (1.0f + __expf(-x));
}

// ---------------------------------------------------------------------------
// Prep kernels
// ---------------------------------------------------------------------------

// msg_W [202 x 1212] f32 -> Bt [1248 x 224] bf16 (Bt[c][k] = W[k][t*202+j])
__global__ void prep_msgW(const float* __restrict__ W, bf16_t* __restrict__ Bt) {
  int idx = blockIdx.x * blockDim.x + threadIdx.x;
  if (idx >= PCOLS * HP) return;
  int c = idx / HP, k = idx % HP;
  int t = c / CHUNK, j = c % CHUNK;
  float v = (j < HH && k < HH) ? W[(size_t)k * (ETYPES * HH) + t * HH + j] : 0.0f;
  Bt[idx] = (bf16_t)v;
}

// gru_W [606 x 202] f32 -> Bt [608 x 224] bf16 (Bt[j][k] = W[j][k], i.e. B=W^T)
__global__ void prep_gruW(const float* __restrict__ W, bf16_t* __restrict__ Bt) {
  int idx = blockIdx.x * blockDim.x + threadIdx.x;
  if (idx >= GCOLS * HP) return;
  int jc = idx / HP, k = idx % HP;
  float v = (jc < 3 * HH && k < HH) ? W[(size_t)jc * HH + k] : 0.0f;
  Bt[idx] = (bf16_t)v;
}

// gate[p][j] = 2*sigmoid(pos_table[p] . pos_W[:,j] + pos_b[j]), stride HP
__global__ void gate_kernel(const float* __restrict__ posW,
                            const float* __restrict__ posb,
                            float* __restrict__ gate) {
  __shared__ float tab[HH];
  int p = blockIdx.x;
  for (int k = threadIdx.x; k < HH; k += blockDim.x) {
    int ki    = (k < 101) ? k : (k - 101);
    float invf = __expf(-((2.0f * (float)ki) / (float)HH) * logf(10000.0f));
    float ang  = (float)p * invf;
    tab[k] = (k < 101) ? __sinf(ang) : __cosf(ang);
  }
  __syncthreads();
  for (int j = threadIdx.x; j < HP; j += blockDim.x) {
    float s = 0.0f;
    if (j < HH) {
      s = posb[j];
      for (int k = 0; k < HH; ++k) s += tab[k] * posW[(size_t)k * HH + j];
      s = 2.0f * sigmoidf_(s);
    }
    gate[(size_t)p * HP + j] = s;
  }
}

__global__ void zero_f32(float* __restrict__ p, size_t n) {
  size_t i = (size_t)blockIdx.x * blockDim.x + threadIdx.x;
  size_t stride = (size_t)gridDim.x * blockDim.x;
  for (; i < n; i += stride) p[i] = 0.0f;
}

__global__ void init_h(const float* __restrict__ ns,
                       float* __restrict__ hf, bf16_t* __restrict__ hbf) {
  size_t idx = (size_t)blockIdx.x * blockDim.x + threadIdx.x;
  if (idx >= (size_t)NN * HP) return;
  size_t n = idx / HP, k = idx % HP;
  float v = (k < HH) ? ns[n * HH + k] : 0.0f;
  hf[idx] = v;
  hbf[idx] = (bf16_t)v;
}

__global__ void count_edges(const int* __restrict__ edges, float* __restrict__ cnt) {
  int idx = blockIdx.x * blockDim.x + threadIdx.x;
  if (idx >= ETYPES * EE) return;
  int t = idx / EE, e = idx % EE;
  int ft = t % 3;
  const int* ep = edges + ((size_t)ft * EE + e) * 2;
  int tgt = (t < 3) ? ep[1] : ep[0];
  atomicAdd(&cnt[tgt], 1.0f);
}

__global__ void finalize_div(float* __restrict__ cnt) {
  int i = blockIdx.x * blockDim.x + threadIdx.x;
  if (i >= NN) return;
  float c = cnt[i];
  cnt[i] = (c == 0.0f) ? 1.0f : (1.0f / c);
}

// ---------------------------------------------------------------------------
// GEMM kernels (bf16 WMMA, f32 accumulate)
// ---------------------------------------------------------------------------

// prop[N x 1248] = h_bf16[N x 224] @ msgWt^T, + msg_b per original column
__global__ __launch_bounds__(256)
void gemm_prop(const bf16_t* __restrict__ A, const bf16_t* __restrict__ Bt,
               const float* __restrict__ msgb, float* __restrict__ C) {
  int lane = threadIdx.x & 31;
  int wave = threadIdx.x >> 5;
  int row0 = blockIdx.x * 16;
  int col0 = blockIdx.y * 128 + wave * 16;
  if (col0 >= PCOLS) return;                       // wave-uniform
  v8f acc = {};
  for (int k0 = 0; k0 < HP; k0 += 32) {
    v16bf a = load_a_frag(A, row0, HP, k0, lane);
    v16bf b = load_bt_frag(Bt, col0, HP, k0, lane);
    acc = wmma_bf16(a, b, acc);
  }
  int nCol  = lane & 15;
  int mBase = (lane >> 4) << 3;
  int cg = col0 + nCol;
  int t = cg / CHUNK, j = cg % CHUNK;
  float bias = (j < HH) ? msgb[t * HH + j] : 0.0f;
  for (int i = 0; i < 8; ++i)
    C[(size_t)(row0 + mBase + i) * PCOLS + cg] = acc[i] + bias;
}

// gi = msgs_bf16 @ Wih^T + bih ; gh = h_bf16 @ Whh^T + bhh  (fused A reuse)
__global__ __launch_bounds__(256)
void gemm_gru(const bf16_t* __restrict__ Am, const bf16_t* __restrict__ Ah,
              const bf16_t* __restrict__ BtI, const bf16_t* __restrict__ BtH,
              const float* __restrict__ bih, const float* __restrict__ bhh,
              float* __restrict__ gi, float* __restrict__ gh) {
  int lane = threadIdx.x & 31;
  int wave = threadIdx.x >> 5;
  int row0 = blockIdx.x * 16;
  int col0 = blockIdx.y * 128 + wave * 16;
  if (col0 >= GCOLS) return;                       // wave-uniform
  v8f ai = {}, ah = {};
  for (int k0 = 0; k0 < HP; k0 += 32) {
    v16bf am = load_a_frag(Am, row0, HP, k0, lane);
    v16bf av = load_a_frag(Ah, row0, HP, k0, lane);
    v16bf bi = load_bt_frag(BtI, col0, HP, k0, lane);
    v16bf bh = load_bt_frag(BtH, col0, HP, k0, lane);
    ai = wmma_bf16(am, bi, ai);
    ah = wmma_bf16(av, bh, ah);
  }
  int nCol  = lane & 15;
  int mBase = (lane >> 4) << 3;
  int cg = col0 + nCol;
  float b1 = (cg < 3 * HH) ? bih[cg] : 0.0f;
  float b2 = (cg < 3 * HH) ? bhh[cg] : 0.0f;
  for (int i = 0; i < 8; ++i) {
    size_t off = (size_t)(row0 + mBase + i) * GCOLS + cg;
    gi[off] = ai[i] + b1;
    gh[off] = ah[i] + b2;
  }
}

// ---------------------------------------------------------------------------
// Edge message scatter: one wave per edge
// ---------------------------------------------------------------------------
__global__ __launch_bounds__(256)
void scatter_edges(const float* __restrict__ prop, const float* __restrict__ gate,
                   const int* __restrict__ edges, const int* __restrict__ posl,
                   float* __restrict__ msg) {
  int wave = threadIdx.x >> 5;
  int lane = threadIdx.x & 31;
  long long eidx = (long long)blockIdx.x * 8 + wave;
  if (eidx >= (long long)ETYPES * EE) return;
  int t = (int)(eidx / EE), e = (int)(eidx % EE);
  int ft = t % 3;
  const int* ep = edges + ((size_t)ft * EE + e) * 2;
  int src = (t < 3) ? ep[0] : ep[1];
  int tgt = (t < 3) ? ep[1] : ep[0];
  int pos = posl[(size_t)ft * EE + e];
  pos = min(max(pos, 0), POSMAX - 1);
  const float* pr = prop + (size_t)src * PCOLS + t * CHUNK;
  const float* gr = gate + (size_t)pos * HP;
  float*       mr = msg  + (size_t)tgt * HP;
  for (int j = lane; j < HH; j += 32)
    atomicAdd(&mr[j], pr[j] * gr[j]);
}

// msgs_bf16 = bf16(messages * inv_divisor[n])
__global__ void scale_msg(const float* __restrict__ msg, const float* __restrict__ invd,
                          bf16_t* __restrict__ mbf) {
  size_t idx = (size_t)blockIdx.x * blockDim.x + threadIdx.x;
  if (idx >= (size_t)NN * HP) return;
  size_t n = idx / HP;
  mbf[idx] = (bf16_t)(msg[idx] * invd[n]);
}

// GRU gates elementwise; writes new h (f32 + bf16)
__global__ void gru_elem(const float* __restrict__ gi, const float* __restrict__ gh,
                         float* __restrict__ hf, bf16_t* __restrict__ hbf) {
  size_t idx = (size_t)blockIdx.x * blockDim.x + threadIdx.x;
  if (idx >= (size_t)NN * HH) return;
  size_t n = idx / HH, j = idx % HH;
  size_t b = n * GCOLS;
  float r  = sigmoidf_(gi[b + j] + gh[b + j]);
  float z  = sigmoidf_(gi[b + HH + j] + gh[b + HH + j]);
  float nn = tanhf(gi[b + 2 * HH + j] + r * gh[b + 2 * HH + j]);
  size_t ho = n * HP + j;
  float hnew = (1.0f - z) * nn + z * hf[ho];
  hf[ho] = hnew;
  hbf[ho] = (bf16_t)hnew;
}

// d_out = [h[:, :202] flat, node_states flat]
__global__ void write_out(const float* __restrict__ hf, const float* __restrict__ ns,
                          float* __restrict__ out) {
  size_t idx = (size_t)blockIdx.x * blockDim.x + threadIdx.x;
  if (idx >= (size_t)NN * HH) return;
  size_t n = idx / HH, j = idx % HH;
  out[idx] = hf[n * HP + j];
  out[(size_t)NN * HH + idx] = ns[idx];
}

// ---------------------------------------------------------------------------
// Host launcher
// ---------------------------------------------------------------------------
extern "C" void kernel_launch(void* const* d_in, const int* in_sizes, int n_in,
                              void* d_out, int out_size, void* d_ws, size_t ws_size,
                              hipStream_t stream) {
  (void)in_sizes; (void)n_in; (void)out_size; (void)ws_size;

  const float* node_states = (const float*)d_in[0];
  const int*   edge_lists  = (const int*)  d_in[1];
  const int*   pos_lists   = (const int*)  d_in[2];
  const float* msg_W       = (const float*)d_in[3];
  const float* msg_b       = (const float*)d_in[4];
  const float* pos_W       = (const float*)d_in[5];
  const float* pos_b       = (const float*)d_in[6];
  const float* gru_Wih     = (const float*)d_in[7];
  const float* gru_Whh     = (const float*)d_in[8];
  const float* gru_bih     = (const float*)d_in[9];
  const float* gru_bhh     = (const float*)d_in[10];
  float* out = (float*)d_out;

  // workspace carve-up (256B aligned)
  char* ws = (char*)d_ws;
  size_t off = 0;
  auto carve = [&](size_t bytes) -> char* {
    char* p = ws + off;
    off = (off + bytes + 255) & ~(size_t)255;
    return p;
  };
  bf16_t* hbf    = (bf16_t*)carve((size_t)NN * HP * sizeof(bf16_t));
  float*  hf     = (float*) carve((size_t)NN * HP * sizeof(float));
  bf16_t* msgWt  = (bf16_t*)carve((size_t)PCOLS * HP * sizeof(bf16_t));
  bf16_t* WihB   = (bf16_t*)carve((size_t)GCOLS * HP * sizeof(bf16_t));
  bf16_t* WhhB   = (bf16_t*)carve((size_t)GCOLS * HP * sizeof(bf16_t));
  float*  gate   = (float*) carve((size_t)POSMAX * HP * sizeof(float));
  float*  invd   = (float*) carve((size_t)NN * sizeof(float));        // counts -> 1/count
  float*  prop   = (float*) carve((size_t)NN * PCOLS * sizeof(float));
  float*  msg    = (float*) carve((size_t)NN * HP * sizeof(float));
  bf16_t* msgbf  = (bf16_t*)carve((size_t)NN * HP * sizeof(bf16_t));
  float*  gi     = (float*) carve((size_t)NN * GCOLS * sizeof(float));
  float*  gh     = (float*) carve((size_t)NN * GCOLS * sizeof(float));

  const int B = 256;
  const size_t nHP = (size_t)NN * HP;         // 11.2M
  const size_t nHH = (size_t)NN * HH;         // 10.1M

  // ---- prep ----
  prep_msgW<<<(PCOLS * HP + B - 1) / B, B, 0, stream>>>(msg_W, msgWt);
  prep_gruW<<<(GCOLS * HP + B - 1) / B, B, 0, stream>>>(gru_Wih, WihB);
  prep_gruW<<<(GCOLS * HP + B - 1) / B, B, 0, stream>>>(gru_Whh, WhhB);
  gate_kernel<<<POSMAX, B, 0, stream>>>(pos_W, pos_b, gate);
  zero_f32<<<1024, B, 0, stream>>>(invd, (size_t)NN);
  count_edges<<<(ETYPES * EE + B - 1) / B, B, 0, stream>>>(edge_lists, invd);
  finalize_div<<<(NN + B - 1) / B, B, 0, stream>>>(invd);
  init_h<<<(int)((nHP + B - 1) / B), B, 0, stream>>>(node_states, hf, hbf);

  // ---- 2 timesteps ----
  dim3 gProp(NN / 16, (PCOLS + 127) / 128);   // (3125, 10)
  dim3 gGru (NN / 16, (GCOLS + 127) / 128);   // (3125, 5)
  for (int t = 0; t < 2; ++t) {
    zero_f32<<<4096, B, 0, stream>>>(msg, nHP);
    gemm_prop<<<gProp, B, 0, stream>>>(hbf, msgWt, msg_b, prop);
    scatter_edges<<<(ETYPES * EE) / 8, B, 0, stream>>>(prop, gate, edge_lists,
                                                       pos_lists, msg);
    scale_msg<<<(int)((nHP + B - 1) / B), B, 0, stream>>>(msg, invd, msgbf);
    gemm_gru<<<gGru, B, 0, stream>>>(msgbf, hbf, WihB, WhhB,
                                     gru_bih, gru_bhh, gi, gh);
    gru_elem<<<(int)((nHH + B - 1) / B), B, 0, stream>>>(gi, gh, hf, hbf);
  }

  // ---- outputs ----
  write_out<<<(int)((nHH + B - 1) / B), B, 0, stream>>>(hf, node_states, out);
}
